// LearnableHypergraph_1176821039600
// MI455X (gfx1250) — matrix operand order repaired and checked by reading, not compiled, not run on hardware
//
#include <hip/hip_runtime.h>
#include <hip/hip_bf16.h>

typedef __attribute__((ext_vector_type(16))) _Float16 v16h;
typedef __attribute__((ext_vector_type(8)))  float    v8f;

#define C_CELLS 4096
#define G_GENES 512
#define HID     128
#define TOP_P   0.9f
#define NGROUPS ((C_CELLS * G_GENES) / 16)   // 131072 groups of 16 elements

// ---------------------------------------------------------------------------
// Phase 1: per-(cell,gene) MLP logits via WMMA, grid-stride over 16-element
// groups. B (w2 broadcast to 16 columns) is wave-invariant and hoisted out of
// the loop; per-group work is only the A build (relu(x*w1+b1) -> f16) plus
// 4 chained V_WMMA_F32_16X16X32_F16 (K = 128). Output written gene-major with
// gumbel fused (TAU == 1).
// ---------------------------------------------------------------------------
__global__ __launch_bounds__(256) void mlp_logits_wmma(
    const float* __restrict__ expr, const float* __restrict__ gn,
    const float* __restrict__ w1, const float* __restrict__ b1,
    const float* __restrict__ w2, const float* __restrict__ b2p,
    float* __restrict__ zT)
{
    __shared__ float s_w1[HID];
    __shared__ float s_b1[HID];
    __shared__ float s_w2[HID];
    if (threadIdx.x < HID) {
        s_w1[threadIdx.x] = w1[threadIdx.x];
        s_b1[threadIdx.x] = b1[threadIdx.x];
        s_w2[threadIdx.x] = w2[threadIdx.x];
    }
    __syncthreads();

    const int lane   = threadIdx.x & 31;
    const int waveId = blockIdx.x * (blockDim.x >> 5) + (threadIdx.x >> 5);
    const int nWaves = gridDim.x * (blockDim.x >> 5);
    const int m      = lane & 15;                 // element slot within group
    const int hi     = lane >> 4;                 // 0: low K-half, 1: high K-half
    const int khalfA = hi * 8;
    const int khalfB = hi * 16;
    const float bias2 = b2p[0];

    // Hoisted B: w2 broadcast to all 16 columns. 32x16 f16 B layout (same as A):
    // lanes0-15: j<8 -> K=j, j>=8 -> K=16+(j-8); lanes16-31 shifted by +8.
    v16h bmat[4];
#pragma unroll
    for (int i = 0; i < 4; ++i) {
#pragma unroll
        for (int j = 0; j < 16; ++j) {
            bmat[i][j] = (_Float16)s_w2[32 * i + khalfA + (j & 7) + ((j >> 3) << 4)];
        }
    }

    for (int grp = waveId; grp < NGROUPS; grp += nWaves) {
        const int base = grp * 16;
        const float x  = expr[base + m];          // lanes L and L+16 share element L

        if (grp + nWaves < NGROUPS) {             // uniform guard, EXEC stays all-1s
            __builtin_prefetch(expr + (base + nWaves * 16) + m, 0, 1);
        }

        v8f acc = {};
#pragma unroll
        for (int i = 0; i < 4; ++i) {             // K = 32*i .. 32*i+31
            v16h a;
#pragma unroll
            for (int j = 0; j < 16; ++j) {
                // 16-bit A 16x32 layout: lanes0-15: j<8 -> K=j, j>=8 -> K=16+(j-8);
                // lanes16-31 shifted by +8 within each half.
                const int ka = 32 * i + khalfA + (j & 7) + ((j >> 3) << 4);
                a[j] = (_Float16)fmaxf(fmaf(x, s_w1[ka], s_b1[ka]), 0.0f);
            }
            acc = __builtin_amdgcn_wmma_f32_16x16x32_f16(
                /*neg_a=*/false, a, /*neg_b=*/false, bmat[i],
                /*c_mod=*/(short)0, acc, /*reuse_a=*/false, /*reuse_b=*/false);
        }

        // C/D 16x16 f32 layout: lanes0-15 hold rows 0..7 in acc[0..7],
        // lanes16-31 hold rows 8..15. Every column is identical.
        const int r = lane & 7;
        const float lg =
            (r < 4) ? ((r < 2) ? ((r == 0) ? acc[0] : acc[1])
                               : ((r == 2) ? acc[2] : acc[3]))
                    : ((r < 6) ? ((r == 4) ? acc[4] : acc[5])
                               : ((r == 6) ? acc[6] : acc[7]));

        if ((lane & 8) == 0) {                    // 16 unique writer lanes
            const int me = base + r + hi * 8;
            const float z = lg + bias2 + gn[me];  // TAU == 1.0
            const int g = me & (G_GENES - 1);
            const int c = me >> 9;                // G_GENES == 512
            zT[g * C_CELLS + c] = z;              // gene-major for phase 2
        }
    }
}

// ---------------------------------------------------------------------------
// Phase 2: one 512-thread block per gene. Softmax over 4096 cells, LDS bitonic
// sort (desc) of (score,cell) keys, cumsum, top-p mask, scatter to cell order.
// ---------------------------------------------------------------------------
__global__ __launch_bounds__(512) void topp_sort(
    const float* __restrict__ zT, float* __restrict__ H)
{
    __shared__ unsigned long long keys[C_CELLS];  // 32 KB
    __shared__ float red[512];                    // 2 KB

    const int g   = blockIdx.x;
    const int tid = threadIdx.x;

    // ---- coalesced column load + running max ----
    float loc[8];
    float mx = -3.402823466e38f;
#pragma unroll
    for (int rr = 0; rr < 8; ++rr) {
        const int c = tid + 512 * rr;
        const float v = zT[g * C_CELLS + c];
        loc[rr] = v;
        mx = fmaxf(mx, v);
    }
    red[tid] = mx;
    __syncthreads();
    for (int off = 256; off > 0; off >>= 1) {
        if (tid < off) red[tid] = fmaxf(red[tid], red[tid + off]);
        __syncthreads();
    }
    const float M = red[0];
    __syncthreads();

    // ---- sum of exp ----
    float se = 0.0f;
#pragma unroll
    for (int rr = 0; rr < 8; ++rr) {
        loc[rr] = __expf(loc[rr] - M);
        se += loc[rr];
    }
    red[tid] = se;
    __syncthreads();
    for (int off = 256; off > 0; off >>= 1) {
        if (tid < off) red[tid] += red[tid + off];
        __syncthreads();
    }
    const float inv = 1.0f / red[0];
    __syncthreads();

    // ---- pack (score, cell) keys; scores >= 0 so float bits are monotone ----
#pragma unroll
    for (int rr = 0; rr < 8; ++rr) {
        const int c = tid + 512 * rr;
        const float s = loc[rr] * inv;
        keys[c] = (((unsigned long long)__float_as_uint(s)) << 32) | (unsigned)c;
    }
    __syncthreads();

    // ---- bitonic sort, descending ----
    for (unsigned k = 2; k <= (unsigned)C_CELLS; k <<= 1) {
        for (unsigned j = k >> 1; j > 0; j >>= 1) {
#pragma unroll
            for (int rr = 0; rr < 8; ++rr) {
                const unsigned i   = (unsigned)tid + 512u * rr;
                const unsigned ixj = i ^ j;
                if (ixj > i) {
                    const unsigned long long a = keys[i];
                    const unsigned long long b = keys[ixj];
                    const bool desc = ((i & k) == 0);
                    if (desc ? (a < b) : (a > b)) {
                        keys[i]   = b;
                        keys[ixj] = a;
                    }
                }
            }
            __syncthreads();
        }
    }

    // ---- cumsum: serial within a contiguous 8-chunk, then block scan ----
    float csum[8];
    float tot = 0.0f;
#pragma unroll
    for (int rr = 0; rr < 8; ++rr) {
        const unsigned i = (unsigned)tid * 8u + rr;
        const float s = __uint_as_float((unsigned)(keys[i] >> 32));
        tot += s;
        csum[rr] = tot;
    }
    red[tid] = tot;
    __syncthreads();
    for (int off = 1; off < 512; off <<= 1) {      // Hillis-Steele inclusive
        const float v = (tid >= off) ? red[tid - off] : 0.0f;
        __syncthreads();
        red[tid] += v;
        __syncthreads();
    }
    const float pre = (tid > 0) ? red[tid - 1] : 0.0f;

    // ---- top-p mask + scatter (permutation covers every cell) ----
#pragma unroll
    for (int rr = 0; rr < 8; ++rr) {
        const unsigned i = (unsigned)tid * 8u + rr;
        const float cum = pre + csum[rr];
        // rank 0 always selected: cum is non-decreasing, so if any rank passes
        // cum<=P then rank 0 does too; if none does, rank 0 is forced (ref).
        const bool sel = (cum <= TOP_P) || (i == 0);
        const unsigned cell = (unsigned)(keys[i] & 0xFFFFFFFFull);
        H[cell * G_GENES + g] = sel ? 1.0f : 0.0f;
    }
}

// ---------------------------------------------------------------------------
extern "C" void kernel_launch(void* const* d_in, const int* in_sizes, int n_in,
                              void* d_out, int out_size, void* d_ws, size_t ws_size,
                              hipStream_t stream) {
    (void)in_sizes; (void)n_in; (void)out_size; (void)ws_size;
    const float* expr = (const float*)d_in[0];   // (C,G)
    const float* gn   = (const float*)d_in[1];   // (C,G)
    const float* w1   = (const float*)d_in[2];   // (HID,)
    const float* b1   = (const float*)d_in[3];   // (HID,)
    const float* w2   = (const float*)d_in[4];   // (HID,)
    const float* b2   = (const float*)d_in[5];   // scalar
    float* H  = (float*)d_out;                   // (C,G)
    float* zT = (float*)d_ws;                    // (G,C) gene-major scratch, 8 MB

    // 16384 waves, 8 groups of 16 elements per wave (grid-stride); B hoisted.
    mlp_logits_wmma<<<2048, 256, 0, stream>>>(expr, gn, w1, b1, w2, b2, zT);
    // one block per gene
    topp_sort<<<G_GENES, 512, 0, stream>>>(zT, H);
}
